// Segmentor_25331717112554
// MI455X (gfx1250) — matrix operand order repaired and checked by reading, not compiled, not run on hardware
//
#include <hip/hip_runtime.h>
#include <hip/hip_bf16.h>

// ---------------- problem constants ----------------
#define BQ    8
#define TT    1000
#define TP1   1001
#define DIN0  256
#define HH    256
#define G4H   1024
#define FEAT  512
#define PHI_K 1536
#define NPAD  112     // 100 -> 7 WMMA n-tiles
#define MSEG  50
#define NEGF  (-1000000000.0f)

typedef __attribute__((ext_vector_type(16))) __bf16 v16bf;
typedef __attribute__((ext_vector_type(8)))  float  v8f;

static __device__ __forceinline__ v8f wmma_bf16(v16bf a, v16bf b, v8f c) {
  return __builtin_amdgcn_wmma_f32_16x16x32_bf16(false, a, false, b, (short)0, c,
                                                 false, false);
}

static __device__ __forceinline__ float sigmoidf(float x) {
  return 1.0f / (1.0f + expf(-x));
}

// A-operand swizzle (ISA 7.12.2, 16-bit A 16x32):
// flat layout [mtile][kstep][lane(32)][elem(16)]; per-lane 16 bf16 contiguous.
static __device__ __forceinline__ long a_swz(long R, int k, int Kt) {
  int lr  = (int)(R & 15);
  int k32 = k & 31;
  int g = k32 >> 4, rem = k32 & 15, hf = rem >> 3, vv = (rem & 7) >> 1, odd = k32 & 1;
  int lane = (hf << 4) | lr;
  int elem = (((g << 2) | vv) << 1) | odd;
  return ((((R >> 4) * Kt + (k >> 5)) << 5) + lane) * 16 + elem;
}

// B-operand swizzle (16-bit B 32x16): flat [ntile][kstep][lane][elem]
static __device__ __forceinline__ long b_swz(int nrow, int k, int Kt) {
  int k32 = k & 31, hf = (k32 >> 3) & 1, e = ((k32 >> 4) << 3) | (k32 & 7);
  int lane = (hf << 4) | (nrow & 15);
  return ((((long)(nrow >> 4) * Kt + (k >> 5)) << 5) + lane) * 16 + e;
}

// ---------------- prep kernels ----------------
// src row-major [R][K] f32 -> swizzled bf16 A operand
__global__ void k_packA(const float* __restrict__ src, __bf16* __restrict__ dst,
                        int kshift, int n_total) {
  int idx = blockIdx.x * 256 + threadIdx.x;
  if (idx >= n_total) return;
  int K = 1 << kshift;
  long R = idx >> kshift;
  int k = idx & (K - 1);
  dst[a_swz(R, k, K >> 5)] = (__bf16)src[idx];
}

// W row-major [n=1024][K] f32 -> swizzled bf16 B operand (B[k][n] = W[n][k])
__global__ void k_packB(const float* __restrict__ W, __bf16* __restrict__ dst,
                        int kshift, int n_total) {
  int idx = blockIdx.x * 256 + threadIdx.x;
  if (idx >= n_total) return;
  int K = 1 << kshift;
  int nrow = idx >> kshift;
  int k = idx & (K - 1);
  dst[b_swz(nrow, k, K >> 5)] = (__bf16)W[idx];
}

__global__ void k_add2(const float* __restrict__ a, const float* __restrict__ b,
                       float* __restrict__ dst) {
  int i = blockIdx.x * 256 + threadIdx.x;
  if (i < G4H) dst[i] = a[i] + b[i];
}

// W1 (100,1536) -> chunk-contiguous swizzled B: [kc(12)][nt(7)][ksl(4)][lane][e]
__global__ void k_packW1(const float* __restrict__ W1, __bf16* __restrict__ dst) {
  int idx = blockIdx.x * 256 + threadIdx.x;   // over 112*1536
  if (idx >= NPAD * PHI_K) return;
  int n = idx / PHI_K, k = idx - n * PHI_K;
  float v = (n < 100) ? W1[(long)n * PHI_K + k] : 0.0f;
  int ksg = k >> 5, kc = ksg >> 2, ksl = ksg & 3;
  int k32 = k & 31, hf = (k32 >> 3) & 1, e = ((k32 >> 4) << 3) | (k32 & 7);
  int lane = (hf << 4) | (n & 15), nt = n >> 4;
  long o = ((((long)(kc * 7 + nt) * 4 + ksl) * 32 + lane) << 4) + e;
  dst[o] = (__bf16)v;
}

__global__ void k_w2norm(const float* __restrict__ W2, float* __restrict__ out) {
  if (threadIdx.x == 0 && blockIdx.x == 0) {
    float s = 0.0f;
    for (int n = 0; n < 100; ++n) s += W2[n] * W2[n];
    out[0] = sqrtf(s);
  }
}

// ---------------- input-projection GEMM (WMMA, swizzled operands) ----------------
__global__ __launch_bounds__(256) void k_gemm_in(
    const __bf16* __restrict__ A, const __bf16* __restrict__ Bw,
    const float* __restrict__ bsum, float* __restrict__ gates, int K) {
  int wave = threadIdx.x >> 5, lane = threadIdx.x & 31;
  int half = lane >> 4, lr = lane & 15;
  int tile = blockIdx.x * 8 + wave;         // 500*64 tiles
  int mtile = tile >> 6, ntile = tile & 63;
  int Kt = K >> 5;
  const __bf16* Ab = A + (((long)mtile * Kt) << 9) + lane * 16;
  const __bf16* Bb = Bw + (((long)ntile * Kt) << 9) + lane * 16;
  v8f acc = {};
  for (int ks = 0; ks < Kt; ++ks) {
    v16bf af  = *(const v16bf*)(Ab + ((long)ks << 9));
    v16bf bfm = *(const v16bf*)(Bb + ((long)ks << 9));
    acc = wmma_bf16(af, bfm, acc);
  }
  int col = (ntile << 4) + lr;
#pragma unroll
  for (int r = 0; r < 8; ++r) {
    int grow = mtile * 16 + r + (half << 3);
    gates[((long)grow << 10) + col] = acc[r] + bsum[col];
  }
}

// ---------------- LSTM recurrence (1 WG / direction, WMMA h@Wh^T) ----------------
__global__ __launch_bounds__(1024) void k_lstm_rec(
    const float* __restrict__ gates_f, const float* __restrict__ gates_b,
    const __bf16* __restrict__ Whf, const __bf16* __restrict__ Whb,
    float* __restrict__ out, __bf16* __restrict__ out_bf) {
  __shared__ __bf16 hsw[8 * 32 * 16];   // swizzled A operand (rows 8..15 stay 0)
  __shared__ float  g_pre[BQ][G4H];
  __shared__ float  c_st[BQ][HH];
  int dir = blockIdx.x;
  const float*  gates = dir ? gates_b : gates_f;
  const __bf16* Wh    = dir ? Whb : Whf;
  int tid = threadIdx.x, wave = tid >> 5, lane = tid & 31;
  int half = lane >> 4, lr = lane & 15;
  int n1 = wave * 2, n2 = wave * 2 + 1;
  int col1 = (n1 << 4) + lr, col2 = (n2 << 4) + lr;
  const __bf16* Bb1 = Wh + (((long)n1 * 8) << 9) + lane * 16;
  const __bf16* Bb2 = Wh + (((long)n2 * 8) << 9) + lane * 16;
  for (int i = tid; i < 8 * 32 * 16; i += 1024) hsw[i] = (__bf16)0.0f;
  for (int i = tid; i < BQ * HH; i += 1024) ((float*)c_st)[i] = 0.0f;
  __syncthreads();
  for (int t = 0; t < TT; ++t) {
    int ta = dir ? (TT - 1 - t) : t;
    v8f acc1 = {}, acc2 = {};
#pragma unroll
    for (int ks = 0; ks < 8; ++ks) {
      v16bf af  = *(const v16bf*)&hsw[((ks << 5) + lane) << 4];
      v16bf b1v = *(const v16bf*)(Bb1 + ((long)ks << 9));
      v16bf b2v = *(const v16bf*)(Bb2 + ((long)ks << 9));
      acc1 = wmma_bf16(af, b1v, acc1);
      acc2 = wmma_bf16(af, b2v, acc2);
    }
    if (half == 0) {
#pragma unroll
      for (int r = 0; r < 8; ++r) {   // rows 0..7 == batch
        long gbase = ((long)r * TT + ta) << 10;
        g_pre[r][col1] = acc1[r] + gates[gbase + col1];
        g_pre[r][col2] = acc2[r] + gates[gbase + col2];
      }
    }
    __syncthreads();
#pragma unroll
    for (int q = 0; q < 2; ++q) {
      int item = tid * 2 + q;           // 2048 = 8 batches * 256 dims
      int b = item >> 8, d = item & 255;
      float ig = sigmoidf(g_pre[b][d]);
      float fg = sigmoidf(g_pre[b][256 + d]);
      float gg = tanhf(g_pre[b][512 + d]);
      float og = sigmoidf(g_pre[b][768 + d]);
      float c  = fg * c_st[b][d] + ig * gg;
      float h  = og * tanhf(c);
      c_st[b][d] = c;
      // swizzled store into next step's A operand
      int k32 = d & 31;
      int g2 = k32 >> 4, rem = k32 & 15, hf = rem >> 3, vv = (rem & 7) >> 1, odd = d & 1;
      int lane2 = (hf << 4) | b;
      int elem = (((g2 << 2) | vv) << 1) | odd;
      hsw[((((d >> 5) << 5) + lane2) << 4) + elem] = (__bf16)h;
      long R = (long)b * TT + ta;
      out[R * FEAT + dir * HH + d] = h;
      out_bf[a_swz(R, dir * HH + d, 16)] = (__bf16)h;   // A operand for layer-1 GEMM
    }
    __syncthreads();
  }
}

// ---------------- zero-prepend + cumsum ----------------
__global__ void k_outcum(const float* __restrict__ out1, float* __restrict__ rnn_out,
                         float* __restrict__ rnn_cum) {
  int id = blockIdx.x * 256 + threadIdx.x;   // 4096 = 8*512
  if (id >= BQ * FEAT) return;
  int b = id >> 9, f = id & 511;
  long ob = (long)b * TP1 * FEAT + f;
  rnn_out[ob] = 0.0f;
  rnn_cum[ob] = 0.0f;
  float acc = 0.0f;
  for (int t = 0; t < TT; ++t) {
    float v = out1[((long)b * TT + t) * FEAT + f];
    acc += v;
    rnn_out[ob + (long)(t + 1) * FEAT] = v;
    rnn_cum[ob + (long)(t + 1) * FEAT] = acc;
  }
}

// ---------------- scorer: banded (i,j) MLP via WMMA ------------------------------
// Double-buffered: async-LDS DMA of W1 chunk n+1 and f32 phi-build overlap the
// WMMA work on chunk n.
#define SB_ELEMS (7 * 4 * 32 * 16)    // 14336 bf16 per chunk
#define SA_ELEMS (4 * 4 * 32 * 16)    //  8192 bf16 per chunk
__global__ __launch_bounds__(256) void k_score(
    const float* __restrict__ rnn_out, const float* __restrict__ rnn_cum,
    const __bf16* __restrict__ W1sw, const float* __restrict__ b1,
    const float* __restrict__ a1p, const float* __restrict__ a2p,
    const float* __restrict__ W2, const float* __restrict__ b2,
    const float* __restrict__ w2n, float* __restrict__ sbuf) {
  __shared__ __bf16 sA[2][SA_ELEMS];   // 32 KB, swizzled A (ping-pong)
  __shared__ __bf16 sB[2][SB_ELEMS];   // 56 KB, swizzled B (ping-pong)
  __shared__ float  sH[64][116];
  int b = blockIdx.y, i = blockIdx.x + 1;
  int start = i - MSEG; if (start < 0) start = 0;
  int tid = threadIdx.x, wave = tid >> 5, lane = tid & 31;
  float a1 = a1p[0], a2 = a2p[0];
  const float* cum_i = rnn_cum + ((long)b * TP1 + i) * FEAT;
  const float* out_i = rnn_out + ((long)b * TP1 + i) * FEAT;
  unsigned sB_lds = (unsigned)(size_t)(&sB[0][0]);
  v8f acc[4] = {{}, {}, {}, {}};
  int mt[4], nn[4];
#pragma unroll
  for (int a = 0; a < 4; ++a) {
    int id = a * 8 + wave;
    int idc = (id < 28) ? id : 0;    // keep EXEC uniform around WMMA
    mt[a] = idc / 7; nn[a] = idc % 7;
  }
  // async DMA of one W1 chunk (linear in swizzled layout) into LDS buffer `buf`
  auto stageB = [&](int buf, int kcc) {
    const __bf16* gsrc = W1sw + (long)kcc * SB_ELEMS;
    unsigned base = sB_lds + (unsigned)buf * (SB_ELEMS * 2);
    for (int q = tid; q < SB_ELEMS / 8; q += 256) {   // 1792 x b128
      unsigned lds_a = base + (unsigned)q * 16u;
      unsigned g_off = (unsigned)q * 16u;
      asm volatile("global_load_async_to_lds_b128 %0, %1, %2"
                   :: "v"(lds_a), "v"(g_off), "s"(gsrc) : "memory");
    }
  };
  // build prelu(phi) for one chunk in f32, swizzled bf16 stores
  auto buildA = [&](int buf, int kcc) {
    for (int idx = tid; idx < 64 * 128; idx += 256) {
      int m = idx >> 7, kk = idx & 127;
      int k0 = (kcc << 7) + kk;
      int j = start + m; int jj = (j > TT) ? TT : j;
      float v;
      if (k0 < FEAT)
        v = cum_i[k0] - rnn_cum[((long)b * TP1 + jj) * FEAT + k0];
      else if (k0 < 2 * FEAT)
        v = rnn_out[((long)b * TP1 + jj) * FEAT + (k0 - FEAT)];
      else
        v = out_i[k0 - 2 * FEAT];
      v = (v >= 0.0f) ? v : a1 * v;
      int mt2 = m >> 4, lr2 = m & 15;
      int ks2 = kk >> 5, k32 = kk & 31;
      int g = k32 >> 4, rem = k32 & 15, hf = rem >> 3, vv = (rem & 7) >> 1, odd = k32 & 1;
      int lane2 = (hf << 4) | lr2;
      int elem = (((g << 2) | vv) << 1) | odd;
      sA[buf][((((mt2 << 2) + ks2) << 5) + lane2) * 16 + elem] = (__bf16)v;
    }
  };
  // prologue: chunk 0 into buffer 0
  stageB(0, 0);
  buildA(0, 0);
  asm volatile("s_wait_asynccnt 0" ::: "memory");
  __syncthreads();
  for (int kc = 0; kc < 12; ++kc) {
    int cur = kc & 1;
    if (kc + 1 < 12) stageB(cur ^ 1, kc + 1);     // DMA overlaps WMMA below
#pragma unroll
    for (int ks = 0; ks < 4; ++ks) {
#pragma unroll
      for (int a = 0; a < 4; ++a) {
        v16bf af  = *(const v16bf*)&sA[cur][((((mt[a] << 2) + ks) << 5) + lane) << 4];
        v16bf bfm = *(const v16bf*)&sB[cur][((((nn[a] << 2) + ks) << 5) + lane) << 4];
        acc[a] = wmma_bf16(af, bfm, acc[a]);
      }
    }
    if (kc + 1 < 12) buildA(cur ^ 1, kc + 1);     // VALU work overlaps matrix pipe
    asm volatile("s_wait_asynccnt 0" ::: "memory");
    __syncthreads();
  }
  int half = lane >> 4, lr = lane & 15;
#pragma unroll
  for (int a = 0; a < 4; ++a) {
    int id = a * 8 + wave;
    if (id < 28) {
      int col = nn[a] * 16 + lr;
#pragma unroll
      for (int r = 0; r < 8; ++r) {
        int row = mt[a] * 16 + r + (half << 3);
        float v = acc[a][r] + ((col < 100) ? b1[col] : 0.0f);
        v = (v >= 0.0f) ? v : a2 * v;
        sH[row][col] = v;
      }
    }
  }
  __syncthreads();
  if (tid < 64) {
    float s = b2[0];
    for (int n = 0; n < 100; ++n) s += sH[tid][n] * W2[n];
    s /= (w2n[0] + 1e-6f);
    sbuf[(((long)b * TT + (i - 1)) << 6) + tid] = s;
  }
}

// ---------------- DP: 8 waves, one per batch ----------------
__global__ __launch_bounds__(256) void k_dp(
    const float* __restrict__ sbuf, const int* __restrict__ length,
    float* __restrict__ best_out, int* __restrict__ prev_out,
    float* __restrict__ final_out) {
  __shared__ float bl[BQ][1008];
  int tid = threadIdx.x, b = tid >> 5, lane = tid & 31;
  for (int idx = lane; idx < 1008; idx += 32) bl[b][idx] = 0.0f;
  __syncthreads();
  if (lane == 0) {
    best_out[b * TP1] = 0.0f;
    prev_out[b * TP1] = 0;
  }
  for (int i = 1; i <= TT; ++i) {
    int start = i - MSEG; if (start < 0) start = 0;
    float bv = NEGF; int bj = start;
#pragma unroll
    for (int q = 0; q < 2; ++q) {
      int m = lane + q * 32;
      int j = start + m;
      bool valid = (m < MSEG) && (j < i);
      float cand = valid ? (bl[b][j] + sbuf[(((long)b * TT + (i - 1)) << 6) + m]) : NEGF;
      if (cand > bv) { bv = cand; bj = j; }
    }
    for (int off = 16; off > 0; off >>= 1) {
      float ov = __shfl_xor(bv, off, 32);
      int   oj = __shfl_xor(bj, off, 32);
      if (ov > bv || (ov == bv && oj < bj)) { bv = ov; bj = oj; }
    }
    if (lane == 0) {
      bl[b][i] = bv;
      best_out[b * TP1 + i] = bv;
      prev_out[b * TP1 + i] = bj;
    }
    __syncthreads();
  }
  if (lane == 0) {
    int L = length[b]; if (L < 0) L = 0; if (L > TT) L = TT;
    final_out[b] = bl[b][L];
  }
}

// ---------------- launcher ----------------
extern "C" void kernel_launch(void* const* d_in, const int* in_sizes, int n_in,
                              void* d_out, int out_size, void* d_ws, size_t ws_size,
                              hipStream_t stream) {
  const float* x      = (const float*)d_in[0];
  const int*   length = (const int*)d_in[1];
  const float* Wi0f = (const float*)d_in[2];
  const float* Wh0f = (const float*)d_in[3];
  const float* bi0f = (const float*)d_in[4];
  const float* bh0f = (const float*)d_in[5];
  const float* Wi0b = (const float*)d_in[6];
  const float* Wh0b = (const float*)d_in[7];
  const float* bi0b = (const float*)d_in[8];
  const float* bh0b = (const float*)d_in[9];
  const float* Wi1f = (const float*)d_in[10];
  const float* Wh1f = (const float*)d_in[11];
  const float* bi1f = (const float*)d_in[12];
  const float* bh1f = (const float*)d_in[13];
  const float* Wi1b = (const float*)d_in[14];
  const float* Wh1b = (const float*)d_in[15];
  const float* bi1b = (const float*)d_in[16];
  const float* bh1b = (const float*)d_in[17];
  const float* a1 = (const float*)d_in[18];
  const float* W1 = (const float*)d_in[19];
  const float* b1 = (const float*)d_in[20];
  const float* a2 = (const float*)d_in[21];
  const float* W2 = (const float*)d_in[22];
  const float* b2 = (const float*)d_in[23];

  char* ws = (char*)d_ws;
  size_t off = 0;
  auto take = [&](size_t bytes) -> size_t {
    size_t o = off; off += (bytes + 255) & ~(size_t)255; return o;
  };
  size_t oXBF   = take((size_t)BQ * TT * DIN0 * 2);
  size_t oWI0F  = take((size_t)DIN0 * G4H * 2);
  size_t oWI0B  = take((size_t)DIN0 * G4H * 2);
  size_t oWI1F  = take((size_t)FEAT * G4H * 2);
  size_t oWI1B  = take((size_t)FEAT * G4H * 2);
  size_t oWH0F  = take((size_t)HH * G4H * 2);
  size_t oWH0B  = take((size_t)HH * G4H * 2);
  size_t oWH1F  = take((size_t)HH * G4H * 2);
  size_t oWH1B  = take((size_t)HH * G4H * 2);
  size_t oBSUM  = take((size_t)4 * G4H * 4);
  size_t oGF    = take((size_t)BQ * TT * G4H * 4);
  size_t oGB    = take((size_t)BQ * TT * G4H * 4);
  size_t oOUT0  = take((size_t)BQ * TT * FEAT * 4);
  size_t oOUT0B = take((size_t)BQ * TT * FEAT * 2);
  size_t oOUT1  = take((size_t)BQ * TT * FEAT * 4);
  size_t oOUT1B = take((size_t)BQ * TT * FEAT * 2);
  size_t oROUT  = take((size_t)BQ * TP1 * FEAT * 4);
  size_t oRCUM  = take((size_t)BQ * TP1 * FEAT * 4);
  size_t oW1BF  = take((size_t)PHI_K * NPAD * 2);
  size_t oW2N   = take(4);
  size_t oSBUF  = take((size_t)BQ * TT * 64 * 4);
  (void)ws_size; (void)in_sizes; (void)n_in; (void)out_size;

  __bf16* XBF   = (__bf16*)(ws + oXBF);
  __bf16* WI0F  = (__bf16*)(ws + oWI0F);
  __bf16* WI0B  = (__bf16*)(ws + oWI0B);
  __bf16* WI1F  = (__bf16*)(ws + oWI1F);
  __bf16* WI1B  = (__bf16*)(ws + oWI1B);
  __bf16* WH0F  = (__bf16*)(ws + oWH0F);
  __bf16* WH0B  = (__bf16*)(ws + oWH0B);
  __bf16* WH1F  = (__bf16*)(ws + oWH1F);
  __bf16* WH1B  = (__bf16*)(ws + oWH1B);
  float*  BSUM  = (float*)(ws + oBSUM);
  float*  GF    = (float*)(ws + oGF);
  float*  GB    = (float*)(ws + oGB);
  float*  OUT0  = (float*)(ws + oOUT0);
  __bf16* OUT0B = (__bf16*)(ws + oOUT0B);
  float*  OUT1  = (float*)(ws + oOUT1);
  __bf16* OUT1B = (__bf16*)(ws + oOUT1B);
  float*  ROUT  = (float*)(ws + oROUT);
  float*  RCUM  = (float*)(ws + oRCUM);
  __bf16* W1SW  = (__bf16*)(ws + oW1BF);
  float*  W2N   = (float*)(ws + oW2N);
  float*  SBUF  = (float*)(ws + oSBUF);

  // ---- prep (pack everything into WMMA fragment layouts) ----
  int nx = BQ * TT * DIN0;
  k_packA<<<(nx + 255) / 256, 256, 0, stream>>>(x, XBF, 8, nx);
  k_packB<<<(G4H * DIN0 + 255) / 256, 256, 0, stream>>>(Wi0f, WI0F, 8, G4H * DIN0);
  k_packB<<<(G4H * DIN0 + 255) / 256, 256, 0, stream>>>(Wi0b, WI0B, 8, G4H * DIN0);
  k_packB<<<(G4H * FEAT + 255) / 256, 256, 0, stream>>>(Wi1f, WI1F, 9, G4H * FEAT);
  k_packB<<<(G4H * FEAT + 255) / 256, 256, 0, stream>>>(Wi1b, WI1B, 9, G4H * FEAT);
  k_packB<<<(G4H * HH + 255) / 256, 256, 0, stream>>>(Wh0f, WH0F, 8, G4H * HH);
  k_packB<<<(G4H * HH + 255) / 256, 256, 0, stream>>>(Wh0b, WH0B, 8, G4H * HH);
  k_packB<<<(G4H * HH + 255) / 256, 256, 0, stream>>>(Wh1f, WH1F, 8, G4H * HH);
  k_packB<<<(G4H * HH + 255) / 256, 256, 0, stream>>>(Wh1b, WH1B, 8, G4H * HH);
  k_add2<<<4, 256, 0, stream>>>(bi0f, bh0f, BSUM + 0 * G4H);
  k_add2<<<4, 256, 0, stream>>>(bi0b, bh0b, BSUM + 1 * G4H);
  k_add2<<<4, 256, 0, stream>>>(bi1f, bh1f, BSUM + 2 * G4H);
  k_add2<<<4, 256, 0, stream>>>(bi1b, bh1b, BSUM + 3 * G4H);
  k_packW1<<<(NPAD * PHI_K + 255) / 256, 256, 0, stream>>>(W1, W1SW);
  k_w2norm<<<1, 1, 0, stream>>>(W2, W2N);

  // ---- layer 0 ----
  k_gemm_in<<<4000, 256, 0, stream>>>(XBF, WI0F, BSUM + 0 * G4H, GF, DIN0);
  k_gemm_in<<<4000, 256, 0, stream>>>(XBF, WI0B, BSUM + 1 * G4H, GB, DIN0);
  k_lstm_rec<<<2, 1024, 0, stream>>>(GF, GB, WH0F, WH0B, OUT0, OUT0B);

  // ---- layer 1 ----
  k_gemm_in<<<4000, 256, 0, stream>>>(OUT0B, WI1F, BSUM + 2 * G4H, GF, FEAT);
  k_gemm_in<<<4000, 256, 0, stream>>>(OUT0B, WI1B, BSUM + 3 * G4H, GB, FEAT);
  k_lstm_rec<<<2, 1024, 0, stream>>>(GF, GB, WH1F, WH1B, OUT1, OUT1B);

  // ---- cumsum + scorer + DP ----
  k_outcum<<<16, 256, 0, stream>>>(OUT1, ROUT, RCUM);
  dim3 gs(TT, BQ);
  k_score<<<gs, 256, 0, stream>>>(ROUT, RCUM, W1SW, b1, a1, a2, W2, b2, W2N, SBUF);

  float* best_out  = (float*)d_out;                    // 8*1001 f32
  int*   prev_out  = (int*)((float*)d_out + BQ * TP1); // 8*1001 int32 region
  float* final_out = (float*)d_out + 2 * BQ * TP1;     // 8 f32
  k_dp<<<1, 256, 0, stream>>>(SBUF, length, best_out, prev_out, final_out);
}
